// LongcatFlashTopkRouter_29300266893621
// MI455X (gfx1250) — compile-verified, matrix operand order checked by reading, not run on hardware
//
#include <hip/hip_runtime.h>
#include <stdint.h>

#define TOKENS 8192
#define HIDDEN 6144
#define NEXP   768
#define TOPK   12
#define SCALEF 2.5f

#define MROWS    32               // tokens per workgroup (2 M-tiles of 16)
#define KC       256              // K-chunk staged in LDS
#define NCHUNK   (HIDDEN / KC)    // 24
#define A_STRIDE (KC + 8)         // ushort stride, +16B pad -> conflict-free ds_load_b128
#define L_STRIDE (NEXP + 4)       // float stride for logits buffer

typedef __attribute__((ext_vector_type(16))) __bf16 v16bf;
typedef __attribute__((ext_vector_type(8)))  float  v8f;

union FragBF {
    v16bf v;
    uint4 q[2];
};

__device__ __forceinline__ unsigned short bf16_rne(float f) {
    unsigned u = __float_as_uint(f);
    u += 0x7FFFu + ((u >> 16) & 1u);          // round-to-nearest-even
    return (unsigned short)(u >> 16);
}
__device__ __forceinline__ float bf16f(unsigned short h) {
    return __uint_as_float(((unsigned)h) << 16);
}
__device__ __forceinline__ void split_bf16(float f, unsigned short& h, unsigned short& l) {
    h = bf16_rne(f);
    l = bf16_rne(f - bf16f(h));
}

// ---------------------------------------------------------------------------
// Kernel 0: W fp32 -> bf16 hi/lo pair in workspace
// ---------------------------------------------------------------------------
__global__ __launch_bounds__(256)
void convert_w_kernel(const float* __restrict__ w,
                      unsigned short* __restrict__ whi,
                      unsigned short* __restrict__ wlo, int n4) {
    int i = blockIdx.x * blockDim.x + threadIdx.x;
    if (i >= n4) return;
    float4 v = reinterpret_cast<const float4*>(w)[i];
    unsigned short h0, h1, h2, h3, l0, l1, l2, l3;
    split_bf16(v.x, h0, l0);
    split_bf16(v.y, h1, l1);
    split_bf16(v.z, h2, l2);
    split_bf16(v.w, h3, l3);
    uint2 hp, lp;
    hp.x = (unsigned)h0 | ((unsigned)h1 << 16);
    hp.y = (unsigned)h2 | ((unsigned)h3 << 16);
    lp.x = (unsigned)l0 | ((unsigned)l1 << 16);
    lp.y = (unsigned)l2 | ((unsigned)l3 << 16);
    reinterpret_cast<uint2*>(whi)[i] = hp;
    reinterpret_cast<uint2*>(wlo)[i] = lp;
}

// ---------------------------------------------------------------------------
// Kernel 1: fused router GEMM (bf16x3 split WMMA) + softmax + biased top-12
// 512 threads = 16 wave32; wave owns 3 N-tiles x 2 M-tiles.
// Double-buffered A staging: next chunk's global loads overlap current GEMM.
// ---------------------------------------------------------------------------
__global__ __launch_bounds__(512)
void router_kernel(const float* __restrict__ x,
                   const unsigned short* __restrict__ whi,
                   const unsigned short* __restrict__ wlo,
                   const float* __restrict__ bias,
                   int* __restrict__ out_idx,
                   float* __restrict__ out_w) {
    __shared__ unsigned short sA_hi[2][MROWS * A_STRIDE];
    __shared__ unsigned short sA_lo[2][MROWS * A_STRIDE];
    __shared__ float          sLog[MROWS * L_STRIDE];

    const int tid  = threadIdx.x;
    const int wave = tid >> 5;                  // 0..15
    const int lane = tid & 31;
    const int t0   = blockIdx.x * MROWS;

    const int lane16 = lane & 15;
    const int koff   = (lane >> 4) ? 8 : 0;     // bf16 WMMA lane-split K offset
    const int n0     = wave * 48;               // 3 N-tiles of 16 per wave

    const v8f vzero = {};
    v8f acc[2][3];
#pragma unroll
    for (int mt = 0; mt < 2; ++mt)
#pragma unroll
        for (int t = 0; t < 3; ++t) acc[mt][t] = vzero;

    // ---- staging identity for this thread
    const int srow = tid >> 4;                  // 32 rows, 16 threads each
    const int sj   = tid & 15;
    const float* xrow = x + (size_t)(t0 + srow) * HIDDEN;

    // ---- prologue: prefetch chunk 0 into registers
    float4 regv[4];
#pragma unroll
    for (int q = 0; q < 4; ++q)
        regv[q] = reinterpret_cast<const float4*>(xrow)[sj + q * 16];

    for (int ch = 0; ch < NCHUNK; ++ch) {
        const int cur = ch & 1;

        // ---- convert prefetched registers, store into current buffer
        {
            unsigned short* dh = &sA_hi[cur][srow * A_STRIDE];
            unsigned short* dl = &sA_lo[cur][srow * A_STRIDE];
#pragma unroll
            for (int q = 0; q < 4; ++q) {
                const int fid = sj + q * 16;    // float4 index within row chunk (0..63)
                float4 v = regv[q];
                unsigned short h0, h1, h2, h3, l0, l1, l2, l3;
                split_bf16(v.x, h0, l0);
                split_bf16(v.y, h1, l1);
                split_bf16(v.z, h2, l2);
                split_bf16(v.w, h3, l3);
                uint2 hp, lp;
                hp.x = (unsigned)h0 | ((unsigned)h1 << 16);
                hp.y = (unsigned)h2 | ((unsigned)h3 << 16);
                lp.x = (unsigned)l0 | ((unsigned)l1 << 16);
                lp.y = (unsigned)l2 | ((unsigned)l3 << 16);
                *reinterpret_cast<uint2*>(dh + fid * 4) = hp;
                *reinterpret_cast<uint2*>(dl + fid * 4) = lp;
            }
        }
        __syncthreads();

        // ---- issue next chunk's global loads now; they drain under the WMMAs
        if (ch + 1 < NCHUNK) {
            const float* src = xrow + (size_t)(ch + 1) * KC;
#pragma unroll
            for (int q = 0; q < 4; ++q)
                regv[q] = reinterpret_cast<const float4*>(src)[sj + q * 16];
        }

        // ---- GEMM over this K chunk
        for (int kk = 0; kk < KC / 32; ++kk) {
            FragBF a_hi[2], a_lo[2];
#pragma unroll
            for (int mt = 0; mt < 2; ++mt) {
                const int arow = mt * 16 + lane16;
                const unsigned short* ah = &sA_hi[cur][arow * A_STRIDE + kk * 32 + koff];
                const unsigned short* al = &sA_lo[cur][arow * A_STRIDE + kk * 32 + koff];
                a_hi[mt].q[0] = *reinterpret_cast<const uint4*>(ah);
                a_hi[mt].q[1] = *reinterpret_cast<const uint4*>(ah + 16);
                a_lo[mt].q[0] = *reinterpret_cast<const uint4*>(al);
                a_lo[mt].q[1] = *reinterpret_cast<const uint4*>(al + 16);
            }

            const size_t kglob = (size_t)ch * KC + (size_t)kk * 32 + koff;
#pragma unroll
            for (int t = 0; t < 3; ++t) {
                const size_t boff = (size_t)(n0 + t * 16 + lane16) * HIDDEN + kglob;
                FragBF b_hi, b_lo;
                b_hi.q[0] = *reinterpret_cast<const uint4*>(whi + boff);
                b_hi.q[1] = *reinterpret_cast<const uint4*>(whi + boff + 16);
                b_lo.q[0] = *reinterpret_cast<const uint4*>(wlo + boff);
                b_lo.q[1] = *reinterpret_cast<const uint4*>(wlo + boff + 16);
#pragma unroll
                for (int mt = 0; mt < 2; ++mt) {
                    acc[mt][t] = __builtin_amdgcn_wmma_f32_16x16x32_bf16(
                        false, a_hi[mt].v, false, b_hi.v, (short)0, acc[mt][t], false, false);
                    acc[mt][t] = __builtin_amdgcn_wmma_f32_16x16x32_bf16(
                        false, a_hi[mt].v, false, b_lo.v, (short)0, acc[mt][t], false, false);
                    acc[mt][t] = __builtin_amdgcn_wmma_f32_16x16x32_bf16(
                        false, a_lo[mt].v, false, b_hi.v, (short)0, acc[mt][t], false, false);
                }
            }
        }
    }

    // ---- spill logits to LDS in token-major layout
    const int mbase = (lane >> 4) << 3;   // C/D layout: lanes 16-31 hold rows 8..15
#pragma unroll
    for (int mt = 0; mt < 2; ++mt) {
#pragma unroll
        for (int t = 0; t < 3; ++t) {
            const int col = n0 + t * 16 + lane16;
#pragma unroll
            for (int r = 0; r < 8; ++r)
                sLog[(mt * 16 + mbase + r) * L_STRIDE + col] = acc[mt][t][r];
        }
    }
    __syncthreads();

    // ---- softmax + bias-corrected top-12 (one wave per token, 2 tokens/wave)
    for (int it = 0; it < 2; ++it) {
        const int m = wave * 2 + it;
        const float* lrow = sLog + m * L_STRIDE;

        float l[24];
        float mx = -__builtin_inff();
#pragma unroll
        for (int i = 0; i < 24; ++i) {
            l[i] = lrow[lane + 32 * i];
            mx = fmaxf(mx, l[i]);
        }
#pragma unroll
        for (int off = 16; off >= 1; off >>= 1)
            mx = fmaxf(mx, __shfl_xor(mx, off, 32));

        float sm = 0.f;
#pragma unroll
        for (int i = 0; i < 24; ++i) {
            l[i] = __expf(l[i] - mx);
            sm += l[i];
        }
#pragma unroll
        for (int off = 16; off >= 1; off >>= 1)
            sm += __shfl_xor(sm, off, 32);
        const float inv = 1.0f / sm;

        float s[24], c[24];
#pragma unroll
        for (int i = 0; i < 24; ++i) {
            s[i] = l[i] * inv;                       // softmax score
            c[i] = s[i] + bias[lane + 32 * i];       // selection value
        }

        for (int k = 0; k < TOPK; ++k) {
            float bv = c[0];
            int   be = lane;
            float bs = s[0];
#pragma unroll
            for (int i = 1; i < 24; ++i) {
                if (c[i] > bv) { bv = c[i]; be = lane + 32 * i; bs = s[i]; }
            }
#pragma unroll
            for (int off = 16; off >= 1; off >>= 1) {
                float ov = __shfl_xor(bv, off, 32);
                int   oe = __shfl_xor(be, off, 32);
                float os = __shfl_xor(bs, off, 32);
                if (ov > bv || (ov == bv && oe < be)) { bv = ov; be = oe; bs = os; }
            }
            if (lane == 0) {
                out_idx[(size_t)(t0 + m) * TOPK + k] = be;
                out_w [(size_t)(t0 + m) * TOPK + k] = bs * SCALEF;
            }
#pragma unroll
            for (int i = 0; i < 24; ++i) {
                if (be == lane + 32 * i) c[i] = -__builtin_inff();
            }
        }
    }
}

// ---------------------------------------------------------------------------
extern "C" void kernel_launch(void* const* d_in, const int* in_sizes, int n_in,
                              void* d_out, int out_size, void* d_ws, size_t ws_size,
                              hipStream_t stream) {
    (void)in_sizes; (void)n_in; (void)out_size; (void)ws_size;

    const float* x    = (const float*)d_in[0];   // [8192, 6144]
    const float* w    = (const float*)d_in[1];   // [768, 6144]
    const float* bias = (const float*)d_in[2];   // [768]

    unsigned short* whi = (unsigned short*)d_ws;                     // 9.44 MB
    unsigned short* wlo = whi + (size_t)NEXP * HIDDEN;               // 9.44 MB

    int*   out_idx = (int*)d_out;                                    // [8192,12] int32
    float* out_w   = (float*)((int*)d_out + (size_t)TOKENS * TOPK);  // [8192,12] fp32

    const int n4 = NEXP * HIDDEN / 4;
    convert_w_kernel<<<(n4 + 255) / 256, 256, 0, stream>>>(w, whi, wlo, n4);
    router_kernel<<<TOKENS / MROWS, 512, 0, stream>>>(x, whi, wlo, bias, out_idx, out_w);
}
